// DeepseekV2AttentionMLA_61220463837455
// MI455X (gfx1250) — matrix operand order repaired and checked by reading, not compiled, not run on hardware
//
#include <hip/hip_runtime.h>
#include <hip/hip_bf16.h>

// ---------------- problem constants ----------------
#define B_    32
#define S_    4096
#define HID_  5120
#define H_    16
#define DN_   128
#define DR_   64
#define DV_   128
#define QL_   1536
#define KL_   512
#define CF_   576            // KL_ + DR_
#define QKVN_ 2112
#define QBN_  3072
#define ON_   2048
#define NSPLIT 8
#define SLEN  (S_ / NSPLIT)  // 512
#define NCHUNK (SLEN / 32)   // 16
#define KTT_STRIDE 40        // padded row stride (halves), 80B (16B aligned)
#define SCALE_F 0.07216878364870323f  // (DN_+DR_)^-0.5

typedef __attribute__((ext_vector_type(16))) __bf16 v16bf;
typedef __attribute__((ext_vector_type(8)))  __bf16 v8bf;
typedef __attribute__((ext_vector_type(8)))  float  v8f;

static __device__ inline v8f zero8() {
  v8f z;
#pragma unroll
  for (int i = 0; i < 8; ++i) z[i] = 0.f;
  return z;
}

static __device__ inline v8bf cvt8(float4 f0, float4 f1) {
  v8bf h;
  h[0] = (__bf16)f0.x; h[1] = (__bf16)f0.y; h[2] = (__bf16)f0.z; h[3] = (__bf16)f0.w;
  h[4] = (__bf16)f1.x; h[5] = (__bf16)f1.y; h[6] = (__bf16)f1.z; h[7] = (__bf16)f1.w;
  return h;
}

// A-fragment (16-bit A 16x32): two contiguous 8-half runs per lane.
static __device__ inline v16bf load_a_frag(const __bf16* p, int lane) {
  const int base = (lane >> 4) << 3;
  v8bf lo = *(const v8bf*)(p + base);
  v8bf hi = *(const v8bf*)(p + base + 16);
  return __builtin_shufflevector(lo, hi, 0, 1, 2, 3, 4, 5, 6, 7,
                                 8, 9, 10, 11, 12, 13, 14, 15);
}
// B-fragment (16-bit B 32x16): two contiguous 8-half runs per lane.
static __device__ inline v16bf load_b_frag(const __bf16* p, int lane) {
  const __bf16* q = p + ((lane >> 4) << 3);
  v8bf lo = *(const v8bf*)(q);
  v8bf hi = *(const v8bf*)(q + 16);
  return __builtin_shufflevector(lo, hi, 0, 1, 2, 3, 4, 5, 6, 7,
                                 8, 9, 10, 11, 12, 13, 14, 15);
}

// =====================================================================
// Skinny GEMM: C[32,N] = A[32,K] * B[K,N].  block = 128 (4 waves),
// block covers N=64; DOUBLE-BUFFERED k-tiles: 1 barrier per k-step,
// next tile's global loads overlap current tile's 2 WMMAs.
// grid.x = N/64.
// =====================================================================
__global__ __launch_bounds__(128) void k_gemm32_bf16(
    const float* __restrict__ A, const float* __restrict__ Bm,
    float* __restrict__ C, int N, int K) {
  __shared__ __bf16 As[2][32][32];   // [buf][m][k]
  __shared__ __bf16 Bt[2][64][32];   // [buf][n][k] (transposed)
  const int tid  = threadIdx.x;
  const int lane = tid & 31;
  const int wave = tid >> 5;
  const int bn = blockIdx.x * 64;

  auto stage = [&](int buf, int k0) {
    {
      int r = tid >> 2, c8 = (tid & 3) * 8;
      const float* src = &A[(size_t)r * K + k0 + c8];
      v8bf h = cvt8(*(const float4*)src, *(const float4*)(src + 4));
      *(v8bf*)&As[buf][r][c8] = h;
    }
    for (int i = tid; i < 256; i += 128) {
      int kk = i >> 3, n8 = (i & 7) * 8;
      const float* src = &Bm[(size_t)(k0 + kk) * N + bn + n8];
      v8bf h = cvt8(*(const float4*)src, *(const float4*)(src + 4));
#pragma unroll
      for (int u = 0; u < 8; ++u) Bt[buf][n8 + u][kk] = h[u];
    }
  };

  v8f acc0 = zero8(), acc1 = zero8();
  stage(0, 0);
  __syncthreads();
  for (int k0 = 0; k0 < K; k0 += 32) {
    const int cur = (k0 >> 5) & 1;
    if (k0 + 32 < K) stage(cur ^ 1, k0 + 32);
    v16bf a0 = load_a_frag(&As[cur][lane & 15][0], lane);
    v16bf a1 = load_a_frag(&As[cur][16 + (lane & 15)][0], lane);
    v16bf b  = load_b_frag(&Bt[cur][wave * 16 + (lane & 15)][0], lane);
    acc0 = __builtin_amdgcn_wmma_f32_16x16x32_bf16(false, a0, false, b,
                                                   (short)0, acc0, false, false);
    acc1 = __builtin_amdgcn_wmma_f32_16x16x32_bf16(false, a1, false, b,
                                                   (short)0, acc1, false, false);
    __syncthreads();
  }
  const int n  = bn + wave * 16 + (lane & 15);
  const int m0 = (lane >> 4) << 3;
#pragma unroll
  for (int v = 0; v < 8; ++v) C[(size_t)(m0 + v) * N + n] = acc0[v];
#pragma unroll
  for (int v = 0; v < 8; ++v) C[(size_t)(16 + m0 + v) * N + n] = acc1[v];
}

// =====================================================================
// RMSNorm(q_a), RMSNorm(latent), RoPE(k_pe).  grid = B, block = 256
// =====================================================================
__global__ __launch_bounds__(256) void k_norm_rope(
    const float* __restrict__ qkv, const float* __restrict__ qnw,
    const float* __restrict__ kvnw, const int* __restrict__ pos,
    float* __restrict__ q_a_n, float* __restrict__ latent_n,
    float* __restrict__ k_pe_n) {
  __shared__ float red[256];
  const int b = blockIdx.x, tid = threadIdx.x;
  const float* src = qkv + (size_t)b * QKVN_;

  float s = 0.f;
  for (int i = tid; i < QL_; i += 256) { float x = src[i]; s += x * x; }
  red[tid] = s; __syncthreads();
  for (int o = 128; o > 0; o >>= 1) { if (tid < o) red[tid] += red[tid + o]; __syncthreads(); }
  float r1 = rsqrtf(red[0] / (float)QL_ + 1e-6f);
  __syncthreads();
  for (int i = tid; i < QL_; i += 256)
    q_a_n[(size_t)b * QL_ + i] = src[i] * r1 * qnw[i];

  s = 0.f;
  for (int i = tid; i < KL_; i += 256) { float x = src[QL_ + i]; s += x * x; }
  red[tid] = s; __syncthreads();
  for (int o = 128; o > 0; o >>= 1) { if (tid < o) red[tid] += red[tid + o]; __syncthreads(); }
  float r2 = rsqrtf(red[0] / (float)KL_ + 1e-6f);
  __syncthreads();
  for (int i = tid; i < KL_; i += 256)
    latent_n[(size_t)b * KL_ + i] = src[QL_ + i] * r2 * kvnw[i];

  if (tid < DR_ / 2) {
    float x1 = src[QL_ + KL_ + 2 * tid];
    float x2 = src[QL_ + KL_ + 2 * tid + 1];
    float inv = __powf(10000.f, -(2.f * tid) / (float)DR_);
    float fr = (float)pos[b] * inv;
    float c = __cosf(fr), sn = __sinf(fr);
    k_pe_n[(size_t)b * DR_ + 2 * tid]     = x1 * c - x2 * sn;
    k_pe_n[(size_t)b * DR_ + 2 * tid + 1] = x2 * c + x1 * sn;
  }
}

// =====================================================================
// q prep: rope(q_pe), q_abs = q_nope . w_kc, q_full bf16 (SCALE folded)
// grid = (H, B), block = 128
// =====================================================================
__global__ __launch_bounds__(128) void k_qprep(
    const float* __restrict__ qmat, const int* __restrict__ pos,
    const float* __restrict__ w_kc, __bf16* __restrict__ qful) {
  __shared__ float qn[DN_];
  __shared__ float qp[DR_];
  const int h = blockIdx.x, b = blockIdx.y, tid = threadIdx.x;
  const float* q = qmat + (size_t)b * QBN_ + h * (DN_ + DR_);
  if (tid < DN_) qn[tid] = q[tid];
  if (tid < DR_ / 2) {
    float x1 = q[DN_ + 2 * tid], x2 = q[DN_ + 2 * tid + 1];
    float inv = __powf(10000.f, -(2.f * tid) / (float)DR_);
    float fr = (float)pos[b] * inv;
    float c = __cosf(fr), sn = __sinf(fr);
    qp[2 * tid]     = x1 * c - x2 * sn;
    qp[2 * tid + 1] = x2 * c + x1 * sn;
  }
  __syncthreads();
  __bf16* dst = qful + ((size_t)b * H_ + h) * CF_;
  for (int c0 = 0; c0 < KL_; c0 += 128) {
    int c = c0 + tid;
    float acc = 0.f;
    for (int d = 0; d < DN_; ++d)
      acc += qn[d] * w_kc[((size_t)h * DN_ + d) * KL_ + c];
    dst[c] = (__bf16)(acc * SCALE_F);
  }
  if (tid < DR_) dst[KL_ + tid] = (__bf16)(qp[tid] * SCALE_F);
}

// =====================================================================
// Fused flash-decode, DOUBLE-BUFFERED cache tiles.
// Dynamic LDS: qf 18432 | kt[2] 73728 | ktT[2] 81920 | sc 2048 |
// pbuf 1024 | pm/pl/pscale 192  => 177344 B (< 320KB WGP LDS)
// Chunk it+1's global loads are issued before chunk it's WMMAs.
// =====================================================================
#define FLASH_LDS_BYTES 177344

__global__ __launch_bounds__(128) void k_mla_flash(
    const float* __restrict__ lat, const float* __restrict__ rop,
    const __bf16* __restrict__ qful, const float* __restrict__ latent_n,
    const float* __restrict__ k_pe_n, float* __restrict__ pctx,
    float* __restrict__ pm_g, float* __restrict__ pl_g) {
  extern __shared__ char smem[];
  __bf16* qf   = (__bf16*)smem;                        // [16][576]
  __bf16* kt0  = qf + H_ * CF_;                        // [2][32][576]
  __bf16* ktT0 = kt0 + 2 * 32 * CF_;                   // [2][512][KTT_STRIDE]
  float*  sc   = (float*)(ktT0 + 2 * KL_ * KTT_STRIDE);// [16][32]
  __bf16* pbuf = (__bf16*)(sc + H_ * 32);              // [16][32]
  float*  pm   = (float*)(pbuf + H_ * 32);
  float*  pl   = pm + H_;
  float*  pscale = pl + H_;

  const int p = blockIdx.x, b = blockIdx.y;
  const int tid = threadIdx.x, lane = tid & 31, wave = tid >> 5;
  const int sbase = p * SLEN;

  auto stage = [&](int buf, int s0) {
    __bf16* ktb  = kt0 + buf * (32 * CF_);
    __bf16* ktTb = ktT0 + buf * (KL_ * KTT_STRIDE);
    for (int i = tid; i < 32 * (CF_ / 8); i += 128) {
      int r  = i / (CF_ / 8);
      int c8 = (i % (CF_ / 8)) * 8;
      int s  = s0 + r;
      const float* src;
      if (s == S_ - 1)
        src = (c8 < KL_) ? &latent_n[(size_t)b * KL_ + c8]
                         : &k_pe_n[(size_t)b * DR_ + (c8 - KL_)];
      else
        src = (c8 < KL_) ? &lat[((size_t)b * S_ + s) * KL_ + c8]
                         : &rop[((size_t)b * S_ + s) * DR_ + (c8 - KL_)];
      v8bf h = cvt8(*(const float4*)src, *(const float4*)(src + 4));
      *(v8bf*)&ktb[r * CF_ + c8] = h;
      if (c8 < KL_) {
#pragma unroll
        for (int u = 0; u < 8; ++u) ktTb[(c8 + u) * KTT_STRIDE + r] = h[u];
      }
    }
  };

  // vectorized qf copy
  {
    const uint4* qs = (const uint4*)(qful + (size_t)b * H_ * CF_);
    uint4* qd = (uint4*)qf;
    for (int i = tid; i < H_ * CF_ / 8; i += 128) qd[i] = qs[i];
  }
  if (tid < H_) { pm[tid] = -1e30f; pl[tid] = 0.f; }

  v8f ca[8];
#pragma unroll
  for (int t = 0; t < 8; ++t) ca[t] = zero8();

  stage(0, sbase);
  __syncthreads();

  for (int it = 0; it < NCHUNK; ++it) {
    const int cur = it & 1;
    // ---- issue next chunk's loads first (consumed after end-of-chunk barrier)
    if (it + 1 < NCHUNK) stage(cur ^ 1, sbase + (it + 1) * 32);

    // ---- scores: wave0 -> s[0:16), wave1 -> s[16:32); K=576
    if (wave < 2) {
      v8f sacc = zero8();
      const __bf16* qrow = qf + (lane & 15) * CF_;
      const __bf16* krow = kt0 + cur * (32 * CF_) + (wave * 16 + (lane & 15)) * CF_;
      for (int kk = 0; kk < CF_ / 32; ++kk) {
        v16bf a  = load_a_frag(qrow + kk * 32, lane);
        v16bf bb = load_b_frag(krow + kk * 32, lane);
        sacc = __builtin_amdgcn_wmma_f32_16x16x32_bf16(false, a, false, bb,
                                                       (short)0, sacc, false, false);
      }
      const int m0 = (lane >> 4) << 3;
      const int n  = wave * 16 + (lane & 15);
#pragma unroll
      for (int v = 0; v < 8; ++v) sc[(m0 + v) * 32 + n] = sacc[v];
    }
    __syncthreads();

    // ---- online softmax per head row
    if (tid < H_) {
      const int h = tid;
      float rowmax = -1e30f;
      for (int j = 0; j < 32; ++j) rowmax = fmaxf(rowmax, sc[h * 32 + j]);
      float mnew = fmaxf(pm[h], rowmax);
      float scl  = __expf(pm[h] - mnew);
      float sum = 0.f;
      for (int j = 0; j < 32; ++j) {
        float e = __expf(sc[h * 32 + j] - mnew);
        pbuf[h * 32 + j] = (__bf16)e;
        sum += e;
      }
      pl[h] = pl[h] * scl + sum;
      pm[h] = mnew;
      pscale[h] = scl;
    }
    __syncthreads();

    // ---- ctx += P(16x32) * V(32x128 per wave); B from ktT rows
    {
      const __bf16* ktTb = ktT0 + cur * (KL_ * KTT_STRIDE);
      const int hbase = (lane >> 4) << 3;
      float f[8];
#pragma unroll
      for (int v = 0; v < 8; ++v) f[v] = pscale[hbase + v];
      v16bf a = load_a_frag(pbuf + (lane & 15) * 32, lane);
#pragma unroll
      for (int t = 0; t < 8; ++t) {
#pragma unroll
        for (int v = 0; v < 8; ++v) ca[t][v] *= f[v];
        const int n0 = wave * 128 + t * 16;
        v16bf bb = load_b_frag(ktTb + (n0 + (lane & 15)) * KTT_STRIDE, lane);
        ca[t] = __builtin_amdgcn_wmma_f32_16x16x32_bf16(false, a, false, bb,
                                                        (short)0, ca[t], false, false);
      }
    }
    // end-of-chunk barrier: next buffer fully staged + current buffer free
    __syncthreads();
  }

  // ---- store partials
  {
    const int hb = (lane >> 4) << 3;
#pragma unroll
    for (int t = 0; t < 8; ++t) {
      const int c = wave * 128 + t * 16 + (lane & 15);
#pragma unroll
      for (int v = 0; v < 8; ++v)
        pctx[(((size_t)b * NSPLIT + p) * H_ + hb + v) * KL_ + c] = ca[t][v];
    }
    if (tid < H_) {
      pm_g[((size_t)b * NSPLIT + p) * H_ + tid] = pm[tid];
      pl_g[((size_t)b * NSPLIT + p) * H_ + tid] = pl[tid];
    }
  }
}

// =====================================================================
// Combine split-S partials + v-projection.  grid = (H, B), block = 128
// =====================================================================
__global__ __launch_bounds__(128) void k_combine_vproj(
    const float* __restrict__ pctx, const float* __restrict__ pm_g,
    const float* __restrict__ pl_g, const float* __restrict__ w_vc,
    float* __restrict__ attn_out) {
  __shared__ float ctx[KL_];
  __shared__ float wexp[NSPLIT];
  __shared__ float Lsh;
  const int h = blockIdx.x, b = blockIdx.y, tid = threadIdx.x;

  if (tid == 0) {
    float M = -1e30f;
    for (int p = 0; p < NSPLIT; ++p)
      M = fmaxf(M, pm_g[((size_t)b * NSPLIT + p) * H_ + h]);
    float L = 0.f;
    for (int p = 0; p < NSPLIT; ++p) {
      float w = __expf(pm_g[((size_t)b * NSPLIT + p) * H_ + h] - M);
      wexp[p] = w;
      L += w * pl_g[((size_t)b * NSPLIT + p) * H_ + h];
    }
    Lsh = L;
  }
  __syncthreads();
  const float invL = 1.f / Lsh;
  for (int c = tid; c < KL_; c += 128) {
    float acc = 0.f;
    for (int p = 0; p < NSPLIT; ++p)
      acc += wexp[p] * pctx[(((size_t)b * NSPLIT + p) * H_ + h) * KL_ + c];
    ctx[c] = acc * invL;
  }
  __syncthreads();
  const int v = tid;
  float o = 0.f;
  for (int c = 0; c < KL_; ++c)
    o += ctx[c] * w_vc[((size_t)h * KL_ + c) * DV_ + v];
  attn_out[(size_t)b * ON_ + h * DV_ + v] = o;
}

// =====================================================================
extern "C" void kernel_launch(void* const* d_in, const int* in_sizes, int n_in,
                              void* d_out, int out_size, void* d_ws, size_t ws_size,
                              hipStream_t stream) {
  (void)in_sizes; (void)n_in; (void)out_size; (void)ws_size;
  const float* hidden  = (const float*)d_in[0];
  const int*   pos     = (const int*)d_in[1];
  const float* lat     = (const float*)d_in[2];
  const float* rop     = (const float*)d_in[3];
  const float* w_qkv_a = (const float*)d_in[4];
  const float* qnw     = (const float*)d_in[5];
  const float* w_q_b   = (const float*)d_in[6];
  const float* kvnw    = (const float*)d_in[7];
  const float* w_kc    = (const float*)d_in[8];
  const float* w_vc    = (const float*)d_in[9];
  const float* w_o     = (const float*)d_in[10];
  float* out = (float*)d_out;

  char* ws = (char*)d_ws;
  size_t off = 0;
  auto alloc = [&](size_t bytes) -> void* {
    off = (off + 255) & ~(size_t)255;
    void* p = ws + off;
    off += bytes;
    return p;
  };
  float*  qkv      = (float*)alloc((size_t)B_ * QKVN_ * 4);
  float*  q_a_n    = (float*)alloc((size_t)B_ * QL_ * 4);
  float*  latent_n = (float*)alloc((size_t)B_ * KL_ * 4);
  float*  k_pe_n   = (float*)alloc((size_t)B_ * DR_ * 4);
  float*  qmat     = (float*)alloc((size_t)B_ * QBN_ * 4);
  __bf16* qful     = (__bf16*)alloc((size_t)B_ * H_ * CF_ * 2);
  float*  pctx     = (float*)alloc((size_t)B_ * NSPLIT * H_ * KL_ * 4);
  float*  pm_g     = (float*)alloc((size_t)B_ * NSPLIT * H_ * 4);
  float*  pl_g     = (float*)alloc((size_t)B_ * NSPLIT * H_ * 4);
  float*  attn_out = (float*)alloc((size_t)B_ * ON_ * 4);

  k_gemm32_bf16<<<dim3(QKVN_ / 64), 128, 0, stream>>>(hidden, w_qkv_a, qkv, QKVN_, HID_);
  k_norm_rope<<<dim3(B_), 256, 0, stream>>>(qkv, qnw, kvnw, pos, q_a_n, latent_n, k_pe_n);
  k_gemm32_bf16<<<dim3(QBN_ / 64), 128, 0, stream>>>(q_a_n, w_q_b, qmat, QBN_, QL_);
  k_qprep<<<dim3(H_, B_), 128, 0, stream>>>(qmat, pos, w_kc, qful);
  k_mla_flash<<<dim3(NSPLIT, B_), 128, FLASH_LDS_BYTES, stream>>>(
      lat, rop, qful, latent_n, k_pe_n, pctx, pm_g, pl_g);
  k_combine_vproj<<<dim3(H_, B_), 128, 0, stream>>>(pctx, pm_g, pl_g, w_vc, attn_out);
  k_gemm32_bf16<<<dim3(HID_ / 64), 128, 0, stream>>>(attn_out, w_o, out, HID_, ON_);
}
